// MS_APPT_51539607841
// MI455X (gfx1250) — matrix operand order repaired and checked by reading, not compiled, not run on hardware
//
#include <hip/hip_runtime.h>
#include <hip/hip_bf16.h>
#include <math.h>

// ---------------------------------------------------------------------------
// Shapes (fixed by the reference)
// ---------------------------------------------------------------------------
#define BB   16
#define LL   512
#define EE   1280
#define PP   512
#define FF   256
#define DD   768
#define HH   8
#define DHH  96
#define NEGINF (-1e9f)

#define ACT_NONE 0
#define ACT_RELU 1
#define ACT_TANH 2

typedef __attribute__((ext_vector_type(16))) __bf16 v16bf;
typedef __attribute__((ext_vector_type(8)))  float  v8f;

union BF16Frag {
    unsigned short u[16];
    unsigned       u32[8];
    v16bf          v;
};

// f32 -> bf16 with round-to-nearest-even
__device__ __forceinline__ unsigned short f2bf(float f) {
    union { float f; unsigned u; } x; x.f = f;
    unsigned r = x.u + 0x7FFFu + ((x.u >> 16) & 1u);
    return (unsigned short)(r >> 16);
}
// pack two f32 -> one dword of two bf16
__device__ __forceinline__ unsigned pk2(float x, float y) {
    return (unsigned)f2bf(x) | ((unsigned)f2bf(y) << 16);
}

// ---------------------------------------------------------------------------
// WMMA fragment loaders (CDNA5 16-bit layouts, wave32)
//   A (16x32, MxK): lane l -> m=l&15, half=l>>4; VGPR v -> K = (v&3)*2 + (v>>2)*16 + half*8 (pair)
//   B (32x16, KxN): lane l -> n=l&15, half=l>>4;   VGPR v -> K = (v&3)*2 + (v>>2)*16 + half*8 (pair)
//   C (16x16 f32) : lane l -> n=l&15;              VGPR r -> m = r + (l>>4)*8
// ---------------------------------------------------------------------------
__device__ __forceinline__ v16bf frag_a(const unsigned short* a, int lda, int m0, int k0, int lane) {
    int m = m0 + (lane & 15);
    int half = lane >> 4;
    BF16Frag r;
#pragma unroll
    for (int v = 0; v < 8; ++v) {
        int kb = k0 + (v & 3) * 2 + (v >> 2) * 16 + half * 8;
        r.u32[v] = *(const unsigned*)&a[m * lda + kb];   // two adjacent K values
    }
    return r.v;
}

__device__ __forceinline__ v16bf frag_b(const unsigned short* bm, int ldb, int n0, int k0, int lane) {
    int n = n0 + (lane & 15);
    int half = lane >> 4;
    BF16Frag r;
#pragma unroll
    for (int v = 0; v < 8; ++v) {
        int kb = k0 + (v & 3) * 2 + (v >> 2) * 16 + half * 8;
        r.u[2 * v]     = bm[kb * ldb + n];
        r.u[2 * v + 1] = bm[(kb + 1) * ldb + n];
    }
    return r.v;
}

// B[k][n] = T[n][k]  (gathered transpose, used for Q*K^T)
__device__ __forceinline__ v16bf frag_bT(const unsigned short* t, int ldt, int n0, int k0, int lane) {
    int n = n0 + (lane & 15);
    int half = lane >> 4;
    BF16Frag r;
#pragma unroll
    for (int v = 0; v < 8; ++v) {
        int kb = k0 + (v & 3) * 2 + (v >> 2) * 16 + half * 8;
        r.u32[v] = *(const unsigned*)&t[n * ldt + kb];
    }
    return r.v;
}

// ---------------------------------------------------------------------------
// Wave32 reductions
// ---------------------------------------------------------------------------
__device__ __forceinline__ float wred_sum(float v) {
#pragma unroll
    for (int o = 16; o > 0; o >>= 1) v += __shfl_xor(v, o, 32);
    return v;
}
__device__ __forceinline__ float wred_max(float v) {
#pragma unroll
    for (int o = 16; o > 0; o >>= 1) v = fmaxf(v, __shfl_xor(v, o, 32));
    return v;
}

// ---------------------------------------------------------------------------
// Generic WMMA GEMM:  C[m, coff+n] = act( A_im2col @ W + bias (+ resid) )
//   A logical row r -> (b = r >> lsh, l = r & (L-1)), L = 1<<lsh (8192/512/16,
//   all powers of two). K-tap (dk, c0) tracked incrementally: Kin % 32 == 0 so
//   a 32-wide K tile never straddles a conv tap -> no integer division at all.
//   Plain GEMM: K == Kin (dk stays 0), pad = 0. W row-major [K, N].
//   Tile: 64(M) x 64(N) x 32(K); 256 threads = 8 waves; each wave owns two
//   16x16 accumulators sharing one B fragment -> 2 WMMA / wave / k-step.
// ---------------------------------------------------------------------------
__global__ __launch_bounds__(256)
void gemm_kernel(const float* __restrict__ A, const float* __restrict__ W,
                 const float* __restrict__ bias, const float* __restrict__ resid,
                 float* __restrict__ C,
                 int M, int N, int K, int Kin, int pad, int lsh,
                 int ldc, int coff, int act)
{
    __shared__ unsigned short Ash[64][40];   // 64x32 bf16, padded
    __shared__ unsigned short Bsh[32][72];   // 32x64 bf16, padded

    int tid  = threadIdx.x;
    int lane = tid & 31;
    int w    = tid >> 5;
    int ni   = w & 3;           // N subtile 0..3
    int mh   = w >> 2;          // M half   0..1
    int n0   = blockIdx.x * 64;
    int m0   = blockIdx.y * 64;
    const int Lm   = 1 << lsh;
    const int lmask = Lm - 1;

    v8f acc0 = {}, acc1 = {};
    int dk = 0, c0 = 0;
    for (int k0 = 0; k0 < K; k0 += 32) {
        __syncthreads();
        // stage A tile: 64 rows x 16 bf16-pairs (float2 loads, dword LDS stores)
#pragma unroll
        for (int pe = tid; pe < 64 * 16; pe += 256) {
            int row  = pe >> 4;
            int kp   = (pe & 15) * 2;
            int grow = m0 + row;
            unsigned pv = 0;
            if (grow < M) {
                int b  = grow >> lsh;
                int l  = grow & lmask;
                int sl = l + dk - pad;
                if ((unsigned)sl < (unsigned)Lm) {
                    const float2 f = *(const float2*)&A[(size_t)((b << lsh) + sl) * Kin + c0 + kp];
                    pv = pk2(f.x, f.y);
                }
            }
            *(unsigned*)&Ash[row][kp] = pv;
        }
        // stage B tile: 32 k-rows x 32 bf16-pairs
#pragma unroll
        for (int pe = tid; pe < 32 * 32; pe += 256) {
            int k  = pe >> 5;
            int np = (pe & 31) * 2;
            const float2 f = *(const float2*)&W[(size_t)(k0 + k) * N + n0 + np];
            *(unsigned*)&Bsh[k][np] = pk2(f.x, f.y);
        }
        // prefetch next weight tile into cache (gfx1250 global_prefetch_b8)
        if (k0 + 32 < K)
            __builtin_prefetch(&W[(size_t)(k0 + 32) * N + n0 + (tid & 63)], 0, 3);
        __syncthreads();

        v16bf b  = frag_b(&Bsh[0][0], 72, ni * 16, 0, lane);
        v16bf a0 = frag_a(&Ash[0][0], 40, mh * 32,      0, lane);
        acc0 = __builtin_amdgcn_wmma_f32_16x16x32_bf16(false, a0, false, b, (short)0, acc0, false, false);
        v16bf a1 = frag_a(&Ash[0][0], 40, mh * 32 + 16, 0, lane);
        acc1 = __builtin_amdgcn_wmma_f32_16x16x32_bf16(false, a1, false, b, (short)0, acc1, false, false);

        // advance conv tap (no division; Kin % 32 == 0)
        c0 += 32;
        if (c0 == Kin) { c0 = 0; ++dk; }
    }

    int gn = n0 + ni * 16 + (lane & 15);
#pragma unroll
    for (int half = 0; half < 2; ++half) {
        v8f acc = half ? acc1 : acc0;
        int mb = m0 + mh * 32 + half * 16 + (lane >> 4) * 8;
#pragma unroll
        for (int r = 0; r < 8; ++r) {
            int m = mb + r;
            if (m < M) {
                float v = acc[r];
                if (bias)  v += bias[gn];
                if (resid) v += resid[(size_t)m * ldc + coff + gn];
                if (act == ACT_RELU)      v = fmaxf(v, 0.f);
                else if (act == ACT_TANH) v = tanhf(v);
                C[(size_t)m * ldc + coff + gn] = v;
            }
        }
    }
}

// ---------------------------------------------------------------------------
// LayerNorm (+ optional exact GELU). One block per row. In-place safe.
// ---------------------------------------------------------------------------
__global__ __launch_bounds__(256)
void ln_act_kernel(const float* __restrict__ x, const float* __restrict__ sc,
                   const float* __restrict__ bi, float* __restrict__ y,
                   int Dd, int dogelu)
{
    int row = blockIdx.x, tid = threadIdx.x, lane = tid & 31, wid = tid >> 5;
    const float* xr = x + (size_t)row * Dd;

    float s = 0.f, s2 = 0.f;
    for (int d = tid; d < Dd; d += blockDim.x) { float v = xr[d]; s += v; s2 += v * v; }
    __shared__ float r1[8], r2[8];
    s = wred_sum(s); s2 = wred_sum(s2);
    if (lane == 0) { r1[wid] = s; r2[wid] = s2; }
    __syncthreads();
    int nw = blockDim.x >> 5;
    if (wid == 0) {
        float a = (lane < nw) ? r1[lane] : 0.f;
        float b = (lane < nw) ? r2[lane] : 0.f;
        a = wred_sum(a); b = wred_sum(b);
        if (lane == 0) { r1[0] = a; r2[0] = b; }
    }
    __syncthreads();
    float mean = r1[0] / Dd;
    float var  = r2[0] / Dd - mean * mean;
    float inv  = rsqrtf(var + 1e-5f);
    float* yr = y + (size_t)row * Dd;
    for (int d = tid; d < Dd; d += blockDim.x) {
        float v = (xr[d] - mean) * inv * sc[d] + bi[d];
        if (dogelu) v = 0.5f * v * (1.f + erff(v * 0.70710678f));
        yr[d] = v;
    }
}

// ---------------------------------------------------------------------------
// Fused attention: one block per (b, h, 32-query tile). 128 threads = 4 waves.
// Full 32x512 score row lives in 64KB dynamic LDS (CDNA5 WGP has 320KB).
// Phase 1: S = scale*Q K^T (WMMA, masked) ; Phase 2: softmax (bf16 repack in
// place) ; Phase 3: O = P V (WMMA), divide by row sums, scatter to [B*L, D].
// ---------------------------------------------------------------------------
__global__ __launch_bounds__(128)
void attn_kernel(const float* __restrict__ Q, const float* __restrict__ Kx,
                 const float* __restrict__ V, const float* __restrict__ mask,
                 float* __restrict__ O)
{
    extern __shared__ float Sdyn[];
    float (*Sm)[LL] = (float(*)[LL])Sdyn;            // [32][512] f32 scores
    __shared__ unsigned short Qt[32][104];           // 32x96 bf16 (+pad)
    __shared__ unsigned short Kt[32][104];           // K tile, reused for V tile
    __shared__ float msk[LL];
    __shared__ float rsum[32];

    int tid = threadIdx.x, lane = tid & 31, w = tid >> 5;
    int q0 = blockIdx.x * 32;
    int h  = blockIdx.y;
    int b  = blockIdx.z;
    const size_t rowb = (size_t)b * LL;

    for (int j = tid; j < LL; j += 128) msk[j] = mask[rowb + j];
#pragma unroll
    for (int pe = tid; pe < 32 * 48; pe += 128) {           // float2-pair staging
        int m = pe / 48, d = (pe - m * 48) * 2;
        const float2 f = *(const float2*)&Q[(rowb + q0 + m) * DD + h * DHH + d];
        *(unsigned*)&Qt[m][d] = pk2(f.x, f.y);
    }

    const float scale = 0.10206207f;   // 1/sqrt(96)
    int mi = w & 1, ni = w >> 1;       // phase-1 tile assignment (2x2)

    for (int kt = 0; kt < LL / 32; ++kt) {
        __syncthreads();
#pragma unroll
        for (int pe = tid; pe < 32 * 48; pe += 128) {
            int m = pe / 48, d = (pe - m * 48) * 2;
            const float2 f = *(const float2*)&Kx[(rowb + kt * 32 + m) * DD + h * DHH + d];
            *(unsigned*)&Kt[m][d] = pk2(f.x, f.y);
        }
        __syncthreads();
        v8f acc = {};
#pragma unroll
        for (int kk = 0; kk < DHH; kk += 32) {
            v16bf a  = frag_a (&Qt[0][0], 104, mi * 16, kk, lane);
            v16bf bt = frag_bT(&Kt[0][0], 104, ni * 16, kk, lane);
            acc = __builtin_amdgcn_wmma_f32_16x16x32_bf16(false, a, false, bt, (short)0, acc, false, false);
        }
        int kcol = kt * 32 + ni * 16 + (lane & 15);
        float mval = msk[kcol];
#pragma unroll
        for (int r = 0; r < 8; ++r) {
            int m = mi * 16 + r + (lane >> 4) * 8;
            float sv = acc[r] * scale;
            if (mval == 0.f) sv = NEGINF;
            Sm[m][kcol] = sv;
        }
    }
    __syncthreads();

    // softmax per row; repack probabilities as bf16 in place (write offset
    // 2j bytes always <= read offset 4j bytes within the same row)
    if (tid < 32) {
        int m = tid;
        float mx = -3.4e38f;
        for (int j = 0; j < LL; ++j) mx = fmaxf(mx, Sm[m][j]);
        unsigned short* pr = (unsigned short*)&Sm[m][0];
        float sum = 0.f;
        for (int j = 0; j < LL; ++j) {
            float p = expf(Sm[m][j] - mx);
            sum += p;
            pr[j] = f2bf(p);
        }
        rsum[m] = sum;
    }
    __syncthreads();

    // P @ V : 32x96 output = 2x6 subtiles over 4 waves (3 each)
    int mi2 = w >> 1;
    int nb  = (w & 1) * 3;
    v8f o0 = {}, o1 = {}, o2 = {};
    const unsigned short* Pm = (const unsigned short*)&Sm[0][0];
    const int ldp = LL * 2;   // row stride in ushorts (512 f32 = 1024 ushorts)

    for (int kt = 0; kt < LL / 32; ++kt) {
        __syncthreads();
#pragma unroll
        for (int pe = tid; pe < 32 * 48; pe += 128) {
            int m = pe / 48, d = (pe - m * 48) * 2;
            const float2 f = *(const float2*)&V[(rowb + kt * 32 + m) * DD + h * DHH + d];
            *(unsigned*)&Kt[m][d] = pk2(f.x, f.y);
        }
        __syncthreads();
        v16bf a  = frag_a(Pm, ldp, mi2 * 16, kt * 32, lane);
        v16bf b0 = frag_b(&Kt[0][0], 104, (nb + 0) * 16, 0, lane);
        o0 = __builtin_amdgcn_wmma_f32_16x16x32_bf16(false, a, false, b0, (short)0, o0, false, false);
        v16bf b1 = frag_b(&Kt[0][0], 104, (nb + 1) * 16, 0, lane);
        o1 = __builtin_amdgcn_wmma_f32_16x16x32_bf16(false, a, false, b1, (short)0, o1, false, false);
        v16bf b2 = frag_b(&Kt[0][0], 104, (nb + 2) * 16, 0, lane);
        o2 = __builtin_amdgcn_wmma_f32_16x16x32_bf16(false, a, false, b2, (short)0, o2, false, false);
    }
#pragma unroll
    for (int t = 0; t < 3; ++t) {
        v8f oc = (t == 0) ? o0 : ((t == 1) ? o1 : o2);
        int gn = (nb + t) * 16 + (lane & 15);
#pragma unroll
        for (int r = 0; r < 8; ++r) {
            int m = mi2 * 16 + r + (lane >> 4) * 8;
            O[(rowb + q0 + m) * DD + h * DHH + gn] = oc[r] / rsum[m];
        }
    }
}

// ---------------------------------------------------------------------------
// Pooling-attention score:  s[r] = tanh(x@W1+b1) @ w2 + b2 (masked).
// One wave per row.
// ---------------------------------------------------------------------------
__global__ __launch_bounds__(256)
void score_kernel(const float* __restrict__ t, const float* __restrict__ w2,
                  const float* __restrict__ b2, const float* __restrict__ mask,
                  float* __restrict__ s, int R, int C_)
{
    int row  = blockIdx.x * 8 + (threadIdx.x >> 5);
    int lane = threadIdx.x & 31;
    if (row >= R) return;
    float sum = 0.f;
    for (int c = lane; c < C_; c += 32) sum += t[(size_t)row * C_ + c] * w2[c];
    sum = wred_sum(sum);
    if (lane == 0) {
        float v = sum + b2[0];
        if (mask[row] == 0.f) v = NEGINF;
        s[row] = v;
    }
}

// ---------------------------------------------------------------------------
// Pool: [mean, masked-max, attention] -> f[b, 3*D]. One block per batch.
// ---------------------------------------------------------------------------
__global__ __launch_bounds__(256)
void pool_kernel(const float* __restrict__ x, const float* __restrict__ mask,
                 const float* __restrict__ s, float* __restrict__ f)
{
    int b = blockIdx.x, tid = threadIdx.x, lane = tid & 31, wid = tid >> 5;
    __shared__ float pr[LL];
    __shared__ float mv[LL];
    __shared__ float r1[8];
    __shared__ float stat[2];

    for (int l = tid; l < LL; l += 256) {
        mv[l] = mask[b * LL + l];
        pr[l] = s[b * LL + l];
    }
    __syncthreads();

    float p = 0.f;
    for (int l = tid; l < LL; l += 256) p += mv[l];
    p = wred_sum(p);
    if (lane == 0) r1[wid] = p;
    __syncthreads();
    if (tid == 0) { float t = 0.f; for (int i = 0; i < 8; ++i) t += r1[i]; stat[0] = fmaxf(t, 1.f); }
    __syncthreads();

    float mx = -3.4e38f;
    for (int l = tid; l < LL; l += 256) mx = fmaxf(mx, pr[l]);
    mx = wred_max(mx);
    if (lane == 0) r1[wid] = mx;
    __syncthreads();
    if (tid == 0) { float t = r1[0]; for (int i = 1; i < 8; ++i) t = fmaxf(t, r1[i]); stat[1] = t; }
    __syncthreads();

    float smax = stat[1];
    float e = 0.f;
    for (int l = tid; l < LL; l += 256) { float q = expf(pr[l] - smax); pr[l] = q; e += q; }
    e = wred_sum(e);
    if (lane == 0) r1[wid] = e;
    __syncthreads();
    if (tid == 0) { float t = 0.f; for (int i = 0; i < 8; ++i) t += r1[i]; stat[1] = t; }
    __syncthreads();

    float ssum = stat[1], msum = stat[0];
    for (int l = tid; l < LL; l += 256) pr[l] /= ssum;
    __syncthreads();

    for (int d = tid; d < DD; d += 256) {
        float mean = 0.f, mxv = NEGINF, att = 0.f;
        for (int l = 0; l < LL; ++l) {
            float xv = x[((size_t)b * LL + l) * DD + d];
            mean += xv * mv[l];
            mxv = fmaxf(mxv, (mv[l] > 0.f) ? xv : NEGINF);
            att += xv * pr[l];
        }
        f[b * 3 * DD + d]           = mean / msum;
        f[b * 3 * DD + DD + d]      = mxv;
        f[b * 3 * DD + 2 * DD + d]  = att;
    }
}

// pair = [f1, f2, |f1-f2|, f1*f2]
__global__ __launch_bounds__(256)
void pair_kernel(const float* __restrict__ f1, const float* __restrict__ f2,
                 float* __restrict__ pair)
{
    int b = blockIdx.x;
    float* pb = pair + (size_t)b * 4 * 3 * DD;
    for (int j = threadIdx.x; j < 3 * DD; j += blockDim.x) {
        float a = f1[b * 3 * DD + j], c = f2[b * 3 * DD + j];
        pb[j]               = a;
        pb[3 * DD + j]      = c;
        pb[2 * 3 * DD + j]  = fabsf(a - c);
        pb[3 * 3 * DD + j]  = a * c;
    }
}

// final N=1 head: out[row] = x[row,:] . w + b
__global__ __launch_bounds__(32)
void head_kernel(const float* __restrict__ x, const float* __restrict__ w,
                 const float* __restrict__ bi, float* __restrict__ out, int C_)
{
    int row = blockIdx.x, lane = threadIdx.x;
    float sum = 0.f;
    for (int c = lane; c < C_; c += 32) sum += x[(size_t)row * C_ + c] * w[c];
    sum = wred_sum(sum);
    if (lane == 0) out[row] = sum + bi[0];
}

// ---------------------------------------------------------------------------
// Host orchestration
// ---------------------------------------------------------------------------
extern "C" void kernel_launch(void* const* d_in, const int* in_sizes, int n_in,
                              void* d_out, int out_size, void* d_ws, size_t ws_size,
                              hipStream_t stream)
{
    (void)in_sizes; (void)n_in; (void)out_size; (void)ws_size;

    // Input order mirrors setup_inputs() dict insertion order, leaves flattened
    const float* emb1   = (const float*)d_in[0];
    const float* emb2   = (const float*)d_in[1];
    const float* mask1  = (const float*)d_in[2];
    const float* mask2  = (const float*)d_in[3];
    const float* proj_w = (const float*)d_in[4];
    const float* proj_b = (const float*)d_in[5];
    const float* conv_w[3] = { (const float*)d_in[6], (const float*)d_in[7], (const float*)d_in[8] };
    const float* conv_b[3] = { (const float*)d_in[9], (const float*)d_in[10], (const float*)d_in[11] };
    const int SELF0 = 12, SELF1 = 22, CROSS0 = 32, CROSS1 = 42;   // 10 leaves: ln_s,ln_b,wq,bq,wk,bk,wv,bv,wo,bo
    const float* ap_w1 = (const float*)d_in[52];
    const float* ap_b1 = (const float*)d_in[53];
    const float* ap_w2 = (const float*)d_in[54];
    const float* ap_b2 = (const float*)d_in[55];
    const int MLP0 = 56, MLP1 = 60, MLP2 = 64;                    // w,b,ln_s,ln_b each
    const float* out_w = (const float*)d_in[68];
    const float* out_b = (const float*)d_in[69];

    // workspace carve-up (floats); total ~235 MB
    float* ws = (float*)d_ws;
    size_t off = 0;
    auto alloc = [&](size_t n) { float* p = ws + off; off += n; return p; };
    float* HP1 = alloc((size_t)BB * LL * PP);   // relu(proj) seq1; later pooling scratch
    float* HP2 = alloc((size_t)BB * LL * PP);   // relu(proj) seq2; later small buffers
    float* h1  = alloc((size_t)BB * LL * DD);
    float* h2  = alloc((size_t)BB * LL * DD);
    float* h1b = alloc((size_t)BB * LL * DD);
    float* h2b = alloc((size_t)BB * LL * DD);
    float* QN  = alloc((size_t)BB * LL * DD);   // LN output; reused as attention output
    float* QB  = alloc((size_t)BB * LL * DD);
    float* KB  = alloc((size_t)BB * LL * DD);
    float* VB  = alloc((size_t)BB * LL * DD);

    // small buffers overlay HP1/HP2 (conv inputs are dead by pooling time)
    float* STMP = HP1;                       // [8192, 256] tanh(x@W1+b1)
    float* s1   = HP2;                       // [16*512]
    float* s2   = HP2 + BB * LL;
    float* F1   = HP2 + 2 * BB * LL;         // [16, 2304]
    float* F2   = F1 + BB * 3 * DD;
    float* PRB  = F2 + BB * 3 * DD;          // [16, 9216]
    float* M0   = PRB + BB * 4 * 3 * DD;     // [16, 1024]
    float* M1   = M0 + BB * 1024;            // [16, 512]
    float* M2   = M1 + BB * 512;             // [16, 256]

    const int NROWS = BB * LL;               // 8192 (lsh 13)

    auto gemm = [&](const float* A, const float* W, const float* bias, const float* resid,
                    float* C, int M, int N, int K, int Kin, int pad, int lsh,
                    int ldc, int coff, int act) {
        dim3 grid(N / 64, (M + 63) / 64);
        gemm_kernel<<<grid, 256, 0, stream>>>(A, W, bias, resid, C, M, N, K, Kin, pad, lsh, ldc, coff, act);
    };

    auto mha = [&](const float* xq, const float* xkv, const float* kmask,
                   int pbase, float* outbuf, bool selfat) {
        const float* Pp[10];
        for (int i = 0; i < 10; ++i) Pp[i] = (const float*)d_in[pbase + i];
        ln_act_kernel<<<NROWS, 256, 0, stream>>>(xq, Pp[0], Pp[1], QN, DD, 0);
        const float* kv = selfat ? QN : xkv;
        gemm(QN, Pp[2], Pp[3], nullptr, QB, NROWS, DD, DD, DD, 0, 13, DD, 0, ACT_NONE);
        gemm(kv, Pp[4], Pp[5], nullptr, KB, NROWS, DD, DD, DD, 0, 13, DD, 0, ACT_NONE);
        gemm(kv, Pp[6], Pp[7], nullptr, VB, NROWS, DD, DD, DD, 0, 13, DD, 0, ACT_NONE);
        attn_kernel<<<dim3(LL / 32, HH, BB), 128, 32 * LL * sizeof(float), stream>>>(QB, KB, VB, kmask, QN);
        gemm(QN, Pp[8], Pp[9], xq, outbuf, NROWS, DD, DD, DD, 0, 13, DD, 0, ACT_NONE);
    };

    // ---- encoder: proj + relu --------------------------------------------
    gemm(emb1, proj_w, proj_b, nullptr, HP1, NROWS, PP, EE, EE, 0, 13, PP, 0, ACT_RELU);
    gemm(emb2, proj_w, proj_b, nullptr, HP2, NROWS, PP, EE, EE, 0, 13, PP, 0, ACT_RELU);

    // ---- convs as im2col GEMMs (K' = k*512; conv_w[k,P,F] is [k*512, 256])
    const int ks[3] = { 3, 5, 7 };
    for (int i = 0; i < 3; ++i) {
        gemm(HP1, conv_w[i], conv_b[i], nullptr, h1, NROWS, FF, ks[i] * PP, PP, ks[i] / 2, 9, DD, i * FF, ACT_RELU);
        gemm(HP2, conv_w[i], conv_b[i], nullptr, h2, NROWS, FF, ks[i] * PP, PP, ks[i] / 2, 9, DD, i * FF, ACT_RELU);
    }

    // ---- self-attention ---------------------------------------------------
    mha(h1, h1, mask1, SELF0, h1, true);
    mha(h2, h2, mask2, SELF0, h2, true);
    mha(h1, h1, mask1, SELF1, h1, true);
    mha(h2, h2, mask2, SELF1, h2, true);

    // ---- cross-attention (double-buffered) --------------------------------
    mha(h1, h2, mask2, CROSS0, h1b, false);
    mha(h2, h1, mask1, CROSS0, h2b, false);
    { float* t = h1; h1 = h1b; h1b = t; t = h2; h2 = h2b; h2b = t; }
    mha(h1, h2, mask2, CROSS1, h1b, false);
    mha(h2, h1, mask1, CROSS1, h2b, false);
    { float* t = h1; h1 = h1b; h1b = t; t = h2; h2 = h2b; h2b = t; }

    // ---- pooling ----------------------------------------------------------
    gemm(h1, ap_w1, ap_b1, nullptr, STMP, NROWS, 256, DD, DD, 0, 13, 256, 0, ACT_TANH);
    score_kernel<<<NROWS / 8, 256, 0, stream>>>(STMP, ap_w2, ap_b2, mask1, s1, NROWS, 256);
    pool_kernel<<<BB, 256, 0, stream>>>(h1, mask1, s1, F1);

    gemm(h2, ap_w1, ap_b1, nullptr, STMP, NROWS, 256, DD, DD, 0, 13, 256, 0, ACT_TANH);
    score_kernel<<<NROWS / 8, 256, 0, stream>>>(STMP, ap_w2, ap_b2, mask2, s2, NROWS, 256);
    pool_kernel<<<BB, 256, 0, stream>>>(h2, mask2, s2, F2);

    pair_kernel<<<BB, 256, 0, stream>>>(F1, F2, PRB);

    // ---- MLP head ---------------------------------------------------------
    gemm(PRB, (const float*)d_in[MLP0], (const float*)d_in[MLP0 + 1], nullptr, M0,
         BB, 1024, 4 * 3 * DD, 4 * 3 * DD, 0, 4, BB ? 1024 : 0, 0, ACT_NONE);
    ln_act_kernel<<<BB, 256, 0, stream>>>(M0, (const float*)d_in[MLP0 + 2], (const float*)d_in[MLP0 + 3], M0, 1024, 1);

    gemm(M0, (const float*)d_in[MLP1], (const float*)d_in[MLP1 + 1], nullptr, M1,
         BB, 512, 1024, 1024, 0, 4, 512, 0, ACT_NONE);
    ln_act_kernel<<<BB, 256, 0, stream>>>(M1, (const float*)d_in[MLP1 + 2], (const float*)d_in[MLP1 + 3], M1, 512, 1);

    gemm(M1, (const float*)d_in[MLP2], (const float*)d_in[MLP2 + 1], nullptr, M2,
         BB, 256, 512, 512, 0, 4, 256, 0, ACT_NONE);
    ln_act_kernel<<<BB, 256, 0, stream>>>(M2, (const float*)d_in[MLP2 + 2], (const float*)d_in[MLP2 + 3], M2, 256, 1);

    head_kernel<<<BB, 32, 0, stream>>>(M2, out_w, out_b, (float*)d_out, 256);
}
